// NeRFNetwork_20590073217645
// MI455X (gfx1250) — matrix operand order repaired and checked by reading, not compiled, not run on hardware
//
#include <hip/hip_runtime.h>
#include <math.h>

typedef __attribute__((ext_vector_type(16))) _Float16 v16h;
typedef __attribute__((ext_vector_type(8)))  _Float16 v8h;
typedef __attribute__((ext_vector_type(8)))  float    v8f;

#define DEVINL __device__ __forceinline__

constexpr int THREADS        = 256;          // 8 waves (wave32)
constexpr int ROWS_PER_WAVE  = 32;           // two 16-row WMMA tiles per wave
constexpr int ROWS_PER_BLOCK = 256;          // 8 waves * 32 rows
constexpr int N_TOTAL        = 4096 * 192;   // 786432 samples
constexpr int RGB_ELEMS      = N_TOTAL * 3;

// ---------------- LDS layout (bytes) ----------------
constexpr unsigned SZ_WT128 = 128 * 128 * 2;                  // 32 KB f16
constexpr unsigned OFF_W1T  = 0;
constexpr unsigned OFF_W2T  = OFF_W1T + SZ_WT128;
constexpr unsigned OFF_W3T  = OFF_W2T + SZ_WT128;
constexpr unsigned OFF_WFT  = OFF_W3T + SZ_WT128;
constexpr unsigned OFF_WRT  = OFF_WFT + SZ_WT128;             // 64 x 128 f16
constexpr unsigned OFF_ACTA = OFF_WRT + 64 * 128 * 2;         // 256 x 128 f16
constexpr unsigned OFF_ACTB = OFF_ACTA + 256 * 128 * 2;       // 256 x 128 f16
// actR (256 x 64 f16) aliases each wave's own actB slice (h3 dead by then)
constexpr unsigned OFF_W0   = OFF_ACTB + 256 * 128 * 2;       // 3 x 128 f32
constexpr unsigned OFF_W2X  = OFF_W0  + 3 * 128 * 4;          // 3 x 128 f32
constexpr unsigned OFF_WRX  = OFF_W2X + 3 * 128 * 4;          // 3 x 64  f32
constexpr unsigned OFF_WD   = OFF_WRX + 3 * 64 * 4;           // 128 f32
constexpr unsigned OFF_WO   = OFF_WD  + 128 * 4;              // 64 x 3 f32
constexpr unsigned OFF_B0   = OFF_WO  + 64 * 3 * 4;
constexpr unsigned OFF_B1   = OFF_B0 + 512;
constexpr unsigned OFF_B2   = OFF_B1 + 512;
constexpr unsigned OFF_B3   = OFF_B2 + 512;
constexpr unsigned OFF_BF   = OFF_B3 + 512;
constexpr unsigned OFF_BR   = OFF_BF + 512;                   // 64 f32
constexpr unsigned OFF_BD   = OFF_BR + 256;                   // 1 f32 (padded)
constexpr unsigned OFF_BO   = OFF_BD + 16;                    // 3 f32 (padded)
constexpr unsigned OFF_POS  = OFF_BO + 16;                    // 256 x 3 f32
constexpr unsigned OFF_DIR  = OFF_POS + 256 * 3 * 4;
constexpr unsigned LDS_BYTES = OFF_DIR + 256 * 3 * 4;         // ~286 KB (< 320 KB/WGP)

// Stage weight w[k][n] (f32, row-major K x N) into LDS as f16 transposed [n][k]
// (k contiguous) -> each lane's WMMA B-fragment is 2 x 16-byte contiguous loads.
template<int K, int N>
DEVINL void stage_wt(char* dst, const float* __restrict__ src, int tid) {
  _Float16* d = (_Float16*)dst;
  for (int i = tid; i < K * N; i += THREADS) {
    int n = i / K, k = i % K;                 // K is a power-of-two constant
    d[n * K + k] = (_Float16)src[k * N + n];
  }
}

DEVINL void copy_f32(float* dst, const float* __restrict__ src, int n, int tid) {
  for (int i = tid; i < n; i += THREADS) dst[i] = src[i];
}

// One fused 128 -> (NTILES*16) layer for a single wave's 32-row tile (2 M-tiles).
//  actIn : LDS f16 [32][128] (row stride 128 halves)
//  wt    : LDS f16 [NTILES*16][128] transposed weights
//  bias  : LDS f32 [NTILES*16]
//  extraW: LDS f32 [3][NTILES*16] (concat tail weights) when HAS_EXTRA
//  extraV: LDS f32 [32][3] per-row pos/dir for this wave when HAS_EXTRA
//  actOut: LDS f16 [32][NTILES*16] (relu'd)
template<int NTILES, bool HAS_EXTRA>
DEVINL void layer_wmma(const char* actIn, const char* wt, const float* bias,
                       const float* extraW, const float* extraV,
                       char* actOut, int lane) {
  constexpr int KIN  = 128;
  constexpr int NOUT = NTILES * 16;
  const int col = lane & 15;       // C/D: lane -> column
  const int g   = lane >> 4;       // lane group
  const int m0  = g * 8;           // C/D: VGPR v -> row m0+v (within 16-row tile)

  v8f acc[2][NTILES];
#pragma unroll
  for (int mt = 0; mt < 2; ++mt) {
    float pv[8][3];
    if constexpr (HAS_EXTRA) {
#pragma unroll
      for (int v = 0; v < 8; ++v)
#pragma unroll
        for (int j = 0; j < 3; ++j)
          pv[v][j] = extraV[(mt * 16 + m0 + v) * 3 + j];
    }
#pragma unroll
    for (int nt = 0; nt < NTILES; ++nt) {
      const int n = nt * 16 + col;
      const float c0 = bias[n];
#pragma unroll
      for (int v = 0; v < 8; ++v) acc[mt][nt][v] = c0;
      if constexpr (HAS_EXTRA) {
        const float e0 = extraW[0 * NOUT + n];
        const float e1 = extraW[1 * NOUT + n];
        const float e2 = extraW[2 * NOUT + n];
#pragma unroll
        for (int v = 0; v < 8; ++v)
          acc[mt][nt][v] += pv[v][0] * e0 + pv[v][1] * e1 + pv[v][2] * e2;
      }
    }
  }

#pragma unroll
  for (int kc = 0; kc < KIN / 32; ++kc) {
    // A fragment: row = lane%16; halves 0..7 -> k = kc*32 + 8g + h,
    //             halves 8..15 -> k = kc*32 + 16 + 8g + (h-8)   (ISA 16-bit A 16x32)
    union { v16h v; v8h h[2]; } A0, A1;
    const char* ar = actIn + (col * KIN) * 2 + kc * 64 + g * 16;
    A0.h[0] = *(const v8h*)(ar);
    A0.h[1] = *(const v8h*)(ar + 32);
    const char* ar1 = ar + 16 * KIN * 2;     // second 16-row tile
    A1.h[0] = *(const v8h*)(ar1);
    A1.h[1] = *(const v8h*)(ar1 + 32);
#pragma unroll
    for (int nt = 0; nt < NTILES; ++nt) {
      // B fragment: col = lane%16; halves 0..7 -> k = kc*32 + 8g + h,
      //             halves 8..15 -> k = kc*32 + 16 + 8g + (h-8) (matches A)
      union { v16h v; v8h h[2]; } B;
      const char* bp = wt + ((nt * 16 + col) * KIN) * 2 + kc * 64 + g * 16;
      B.h[0] = *(const v8h*)(bp);
      B.h[1] = *(const v8h*)(bp + 32);
      // each B fragment feeds two WMMAs (both M-tiles)
      acc[0][nt] = __builtin_amdgcn_wmma_f32_16x16x32_f16(
          false, A0.v, false, B.v, (short)0, acc[0][nt], false, false);
      acc[1][nt] = __builtin_amdgcn_wmma_f32_16x16x32_f16(
          false, A1.v, false, B.v, (short)0, acc[1][nt], false, false);
    }
  }

  // epilogue: relu, f16, back to LDS row-major for the next layer's A loads
#pragma unroll
  for (int mt = 0; mt < 2; ++mt) {
#pragma unroll
    for (int nt = 0; nt < NTILES; ++nt) {
      const int n = nt * 16 + col;
#pragma unroll
      for (int v = 0; v < 8; ++v) {
        float f = fmaxf(acc[mt][nt][v], 0.0f);
        ((_Float16*)actOut)[(mt * 16 + m0 + v) * NOUT + n] = (_Float16)f;
      }
    }
  }
}

__global__ __launch_bounds__(THREADS)
void nerf_fused_kernel(const float* __restrict__ x,
                       const float* __restrict__ w0, const float* __restrict__ b0,
                       const float* __restrict__ w1, const float* __restrict__ b1,
                       const float* __restrict__ w2, const float* __restrict__ b2,
                       const float* __restrict__ w3, const float* __restrict__ b3,
                       const float* __restrict__ wd, const float* __restrict__ bd,
                       const float* __restrict__ wf, const float* __restrict__ bf,
                       const float* __restrict__ wr, const float* __restrict__ br,
                       const float* __restrict__ wo, const float* __restrict__ bo,
                       float* __restrict__ out) {
  extern __shared__ char smem[];
  const int tid = threadIdx.x;

  // ---- stage weights into LDS (f16 transposed for WMMA B-fragments) ----
  stage_wt<128, 128>(smem + OFF_W1T, w1, tid);
  stage_wt<128, 128>(smem + OFF_W2T, w2, tid);              // rows 0..127 of (131,128)
  stage_wt<128, 128>(smem + OFF_W3T, w3, tid);
  stage_wt<128, 128>(smem + OFF_WFT, wf, tid);
  stage_wt<128, 64 >(smem + OFF_WRT, wr, tid);              // rows 0..127 of (131,64)
  copy_f32((float*)(smem + OFF_W0),  w0,             384, tid);
  copy_f32((float*)(smem + OFF_W2X), w2 + 128 * 128, 384, tid);  // concat tail
  copy_f32((float*)(smem + OFF_WRX), wr + 128 * 64,  192, tid);  // concat tail
  copy_f32((float*)(smem + OFF_WD),  wd, 128, tid);
  copy_f32((float*)(smem + OFF_WO),  wo, 192, tid);
  copy_f32((float*)(smem + OFF_B0),  b0, 128, tid);
  copy_f32((float*)(smem + OFF_B1),  b1, 128, tid);
  copy_f32((float*)(smem + OFF_B2),  b2, 128, tid);
  copy_f32((float*)(smem + OFF_B3),  b3, 128, tid);
  copy_f32((float*)(smem + OFF_BF),  bf, 128, tid);
  copy_f32((float*)(smem + OFF_BR),  br,  64, tid);
  copy_f32((float*)(smem + OFF_BD),  bd,   1, tid);
  copy_f32((float*)(smem + OFF_BO),  bo,   3, tid);

  // ---- stage input rays: pos / view_dir ----
  const long base = (long)blockIdx.x * ROWS_PER_BLOCK;
  {
    float* POS = (float*)(smem + OFF_POS);
    float* DIR = (float*)(smem + OFF_DIR);
    for (int i = tid; i < ROWS_PER_BLOCK * 6; i += THREADS) {
      int r = i / 6, c = i % 6;
      float v = x[(base + r) * 6 + c];
      if (c < 3) POS[r * 3 + c] = v;
      else       DIR[r * 3 + (c - 3)] = v;
    }
  }
  __syncthreads();

  // ---- layer 0 (3 -> 128), VALU, writes actA ----
  {
    const float* W0  = (const float*)(smem + OFF_W0);
    const float* B0  = (const float*)(smem + OFF_B0);
    const float* POS = (const float*)(smem + OFF_POS);
    _Float16* A = (_Float16*)(smem + OFF_ACTA);
    for (int i = tid; i < ROWS_PER_BLOCK * 128; i += THREADS) {
      int r = i >> 7, n = i & 127;
      float s = B0[n] + POS[r * 3 + 0] * W0[n]
                      + POS[r * 3 + 1] * W0[128 + n]
                      + POS[r * 3 + 2] * W0[256 + n];
      A[r * 128 + n] = (_Float16)fmaxf(s, 0.0f);
    }
  }
  __syncthreads();

  // ---- per-wave fused MLP: each wave privately owns 32 rows ----
  const int wv = tid >> 5, lane = tid & 31;
  char* actA = smem + OFF_ACTA + wv * ROWS_PER_WAVE * 128 * 2;
  char* actB = smem + OFF_ACTB + wv * ROWS_PER_WAVE * 128 * 2;
  char* actR = actB;   // 32 x 64 f16 aliases own actB slice (h3 dead by then)
  const float* myPos = (const float*)(smem + OFF_POS) + wv * ROWS_PER_WAVE * 3;
  const float* myDir = (const float*)(smem + OFF_DIR) + wv * ROWS_PER_WAVE * 3;
  const long sbase = base + wv * ROWS_PER_WAVE;

  // h1 = relu(h0 @ w1 + b1)
  layer_wmma<8, false>(actA, smem + OFF_W1T, (const float*)(smem + OFF_B1),
                       nullptr, nullptr, actB, lane);
  // h2 = relu([h1,pos] @ w2 + b2)  (pos tail folded into C init)
  layer_wmma<8, true >(actB, smem + OFF_W2T, (const float*)(smem + OFF_B2),
                       (const float*)(smem + OFF_W2X), myPos, actA, lane);
  // h3 = relu(h2 @ w3 + b3)
  layer_wmma<8, false>(actA, smem + OFF_W3T, (const float*)(smem + OFF_B3),
                       nullptr, nullptr, actB, lane);

  // density = relu(h3 @ wd + bd): one row per lane, all 32 lanes active
  {
    const _Float16* row = (const _Float16*)actB + lane * 128;
    const float* WD = (const float*)(smem + OFF_WD);
    float s = *(const float*)(smem + OFF_BD);
#pragma unroll 8
    for (int k = 0; k < 128; ++k) s += (float)row[k] * WD[k];
    out[RGB_ELEMS + sbase + lane] = fmaxf(s, 0.0f);
  }

  // features = relu(h3 @ wf + bf)
  layer_wmma<8, false>(actB, smem + OFF_WFT, (const float*)(smem + OFF_BF),
                       nullptr, nullptr, actA, lane);
  // r = relu([features,dir] @ wr + br)  (dir tail folded into C init)
  layer_wmma<4, true >(actA, smem + OFF_WRT, (const float*)(smem + OFF_BR),
                       (const float*)(smem + OFF_WRX), myDir, actR, lane);

  // rgb = sigmoid(r @ wo + bo): 32 rows x 3 channels = 96 = 3 full-wave passes
  {
    const float* WO = (const float*)(smem + OFF_WO);
    const float* BO = (const float*)(smem + OFF_BO);
#pragma unroll
    for (int t = 0; t < 3; ++t) {
      int idx = lane + 32 * t;                 // 0..95, always valid
      int m = idx / 3, c = idx % 3;
      const _Float16* row = (const _Float16*)actR + m * 64;
      float s = BO[c];
#pragma unroll 8
      for (int k = 0; k < 64; ++k) s += (float)row[k] * WO[k * 3 + c];
      out[(sbase + m) * 3 + c] = 1.0f / (1.0f + __expf(-s));
    }
  }
}

extern "C" void kernel_launch(void* const* d_in, const int* in_sizes, int n_in,
                              void* d_out, int out_size, void* d_ws, size_t ws_size,
                              hipStream_t stream) {
  const float* x  = (const float*)d_in[0];
  const float* w0 = (const float*)d_in[1];
  const float* b0 = (const float*)d_in[2];
  const float* w1 = (const float*)d_in[3];
  const float* b1 = (const float*)d_in[4];
  const float* w2 = (const float*)d_in[5];
  const float* b2 = (const float*)d_in[6];
  const float* w3 = (const float*)d_in[7];
  const float* b3 = (const float*)d_in[8];
  const float* wd = (const float*)d_in[9];
  const float* bd = (const float*)d_in[10];
  const float* wf = (const float*)d_in[11];
  const float* bf = (const float*)d_in[12];
  const float* wr = (const float*)d_in[13];
  const float* br = (const float*)d_in[14];
  const float* wo = (const float*)d_in[15];
  const float* bo = (const float*)d_in[16];
  float* out = (float*)d_out;

  dim3 grid(N_TOTAL / ROWS_PER_BLOCK);   // 3072 blocks, 256 samples each
  nerf_fused_kernel<<<grid, THREADS, LDS_BYTES, stream>>>(
      x, w0, b0, w1, b1, w2, b2, w3, b3, wd, bd, wf, bf, wr, br, wo, bo, out);
}
